// DecoderRNN_79525614452822
// MI455X (gfx1250) — matrix operand order repaired and checked by reading, not compiled, not run on hardware
//
#include <hip/hip_runtime.h>

// ---------------------------------------------------------------------------
// Problem dims (match reference)
// ---------------------------------------------------------------------------
#define LDIM 256      // latent
#define HDIM 512      // hidden
#define ODIM 128      // output
#define BDIM 1024     // batch
#define TENC 200      // encoder length

typedef __attribute__((ext_vector_type(16))) __bf16 v16bf;
typedef __attribute__((ext_vector_type(8)))  __bf16 v8bf;
typedef __attribute__((ext_vector_type(8)))  float  v8f;

static __device__ __forceinline__ v8f vzero8() {
  v8f v;
#pragma unroll
  for (int i = 0; i < 8; ++i) v[i] = 0.0f;
  return v;
}

static __device__ __forceinline__ v16bf concat8(v8bf lo, v8bf hi) {
  return __builtin_shufflevector(lo, hi, 0,1,2,3,4,5,6,7,8,9,10,11,12,13,14,15);
}

// A fragment (16x32 bf16) gathered from a row-major bf16 tile in LDS.
// Layout: lanes 0-15 row m=lane, K=kb..kb+15 (kb=0);
// lanes 16-31 same rows, kb=16.  -> two 16B contiguous LDS loads per lane.
static __device__ __forceinline__ v16bf lds_a_frag(const __bf16* base, int stride, int ktile) {
  int lane = threadIdx.x & 31;
  int m    = lane & 15;
  int k0   = (ktile << 5) + ((lane >> 4) << 4);
  const __bf16* p = base + m * stride + k0;
  v8bf lo = *(const v8bf*)p;
  v8bf hi = *(const v8bf*)(p + 8);
  return concat8(lo, hi);
}

// B fragment from fragment-major packed weights: 32 lanes x 16 bf16 contiguous.
static __device__ __forceinline__ v16bf g_b_frag(const __bf16* pack, int frag) {
  int lane = threadIdx.x & 31;
  return *(const v16bf*)(pack + ((size_t)frag << 9) + (lane << 4));
}

static __device__ __forceinline__ v8f wmma_bf16(v16bf a, v16bf b, v8f c) {
  return __builtin_amdgcn_wmma_f32_16x16x32_bf16(false, a, false, b, (short)0, c, false, false);
}

static __device__ __forceinline__ float sigmoidf_(float x) { return 1.0f / (1.0f + __expf(-x)); }
static __device__ __forceinline__ float lrelu_(float x)    { return x > 0.0f ? x : 0.01f * x; }

// ---------------------------------------------------------------------------
// Weight packing: fp32 row-major [N,K] -> bf16 WMMA B-fragments
// frag = ntile*(K/32)+ktile; element: n = ntile*16+(lane&15),
// k = ktile*32 + ((lane>>4)<<4) + i   (contiguous k order, same as lds_a_frag: lane holds one column,
// 16 consecutive K values).
// ---------------------------------------------------------------------------
__global__ void pack_b_kernel(const float* __restrict__ src, __bf16* __restrict__ dst,
                              int N, int K) {
  int total = (N >> 4) * (K >> 5) * 512;
  int idx = blockIdx.x * blockDim.x + threadIdx.x;
  if (idx >= total) return;
  int f = idx >> 9, r = idx & 511;
  int lane = r >> 4, i = r & 15;
  int ktiles = K >> 5;
  int ntile = f / ktiles, ktile = f % ktiles;
  int n = ntile * 16 + (lane & 15);
  int k = ktile * 32 + ((lane >> 4) << 4) + i;
  dst[idx] = (__bf16)src[n * K + k];
}

// W_cat = [W_ih | W_hh] : N=2048, K=1024 (k<512 -> W_ih, else W_hh)
__global__ void pack_wcat_kernel(const float* __restrict__ Wih, const float* __restrict__ Whh,
                                 __bf16* __restrict__ dst) {
  int idx = blockIdx.x * blockDim.x + threadIdx.x;   // 128*32*512 total
  if (idx >= 128 * 32 * 512) return;
  int f = idx >> 9, r = idx & 511;
  int lane = r >> 4, i = r & 15;
  int ntile = f >> 5, ktile = f & 31;
  int n = ntile * 16 + (lane & 15);
  int k = ktile * 32 + ((lane >> 4) << 4) + i;
  float v = (k < HDIM) ? Wih[n * HDIM + k] : Whh[n * HDIM + (k - HDIM)];
  dst[idx] = (__bf16)v;
}

// ---------------------------------------------------------------------------
// Persistent LSTM decoder: 64 WGs x 16 batch rows, 512 threads (16 wave32).
// Wave w owns hidden columns j in [32w, 32w+32) across all 4 gates.
// c stays in VGPRs; h staged in LDS bf16 as next-step A operand.
// ---------------------------------------------------------------------------
__global__ void __launch_bounds__(512, 1)
decoder_kernel(const float* __restrict__ enc_hidden, const float* __restrict__ start_token,
               const float* __restrict__ W_emb, const float* __restrict__ b_emb,
               const float* __restrict__ b_lat, const float* __restrict__ b_lat2,
               const float* __restrict__ b_ih, const float* __restrict__ b_hh,
               const float* __restrict__ b_out,
               const __bf16* __restrict__ wcat, const __bf16* __restrict__ wl2h,
               const __bf16* __restrict__ wl2h2, const __bf16* __restrict__ wout,
               int seq, float* __restrict__ dec_out, float* __restrict__ hT,
               float* __restrict__ cT) {
  __shared__ __align__(32) __bf16 s_act[16 * 2 * HDIM];   // [m][k<512:relu(x) | k>=512:h]
  __shared__ __align__(32) __bf16 s_enc[16 * LDIM];
  __shared__ float s_bias[4 * HDIM];

  const int tid = threadIdx.x;
  const int wave = tid >> 5;
  const int lane = tid & 31;
  const int bbase = blockIdx.x * 16;

  // combined gate bias
  for (int n = tid; n < 4 * HDIM; n += 512) s_bias[n] = b_ih[n] + b_hh[n];

  // x0 = emb(start_token); act x-part = relu(x0), broadcast over the 16 rows
  for (int j = tid; j < HDIM; j += 512) {
    float a = b_emb[j];
#pragma unroll 4
    for (int o = 0; o < ODIM; ++o) a += start_token[o] * W_emb[j * ODIM + o];
    __bf16 rb = (__bf16)(a > 0.0f ? a : 0.0f);
    for (int m = 0; m < 16; ++m) s_act[m * (2 * HDIM) + j] = rb;
  }

  // stage encoder_hidden tile (bf16)
  for (int idx = tid; idx < 16 * LDIM; idx += 512) {
    int m = idx >> 8, k = idx & (LDIM - 1);
    s_enc[idx] = (__bf16)enc_hidden[(bbase + m) * LDIM + k];
  }
  __syncthreads();

  // ---- init: h0/c0 = lrelu(enc_hidden @ W^T + b), WMMA over K=256 ----
  float c_reg[2][8];
  {
    v8f acch[2] = {vzero8(), vzero8()};
    v8f accc[2] = {vzero8(), vzero8()};
    for (int kt = 0; kt < LDIM / 32; ++kt) {
      v16bf a = lds_a_frag(s_enc, LDIM, kt);
#pragma unroll
      for (int u = 0; u < 2; ++u) {
        int frag = (wave * 2 + u) * (LDIM / 32) + kt;
        acch[u] = wmma_bf16(a, g_b_frag(wl2h, frag), acch[u]);
        accc[u] = wmma_bf16(a, g_b_frag(wl2h2, frag), accc[u]);
      }
    }
#pragma unroll
    for (int u = 0; u < 2; ++u) {
      int j = wave * 32 + u * 16 + (lane & 15);
      float bh = b_lat[j], bc = b_lat2[j];
#pragma unroll
      for (int v = 0; v < 8; ++v) {
        int m = v + ((lane >> 4) << 3);
        float h = lrelu_(acch[u][v] + bh);
        c_reg[u][v] = lrelu_(accc[u][v] + bc);
        s_act[m * (2 * HDIM) + HDIM + j] = (__bf16)h;
      }
    }
  }
  __syncthreads();

  // ---- decode loop ----
  for (int t = 0; t < seq; ++t) {
    // gates[16, 2048] = act[16,1024] @ Wcat^T, wave computes its 4x(32-col) slabs
    v8f acc[4][2];
#pragma unroll
    for (int g = 0; g < 4; ++g)
#pragma unroll
      for (int u = 0; u < 2; ++u) acc[g][u] = vzero8();

    for (int kt = 0; kt < 32; ++kt) {
      v16bf a = lds_a_frag(s_act, 2 * HDIM, kt);
#pragma unroll
      for (int g = 0; g < 4; ++g)
#pragma unroll
        for (int u = 0; u < 2; ++u) {
          int ntile = g * 32 + wave * 2 + u;            // n = g*512 + wave*32 + u*16
          acc[g][u] = wmma_bf16(a, g_b_frag(wcat, ntile * 32 + kt), acc[g][u]);
        }
    }
    __syncthreads();   // everyone done reading old act

    // pointwise LSTM cell, update c (regs) and h (LDS bf16 for next A)
    float hnew[2][8];
#pragma unroll
    for (int u = 0; u < 2; ++u) {
      int j = wave * 32 + u * 16 + (lane & 15);
      float bi = s_bias[j], bf_ = s_bias[HDIM + j];
      float bg = s_bias[2 * HDIM + j], bo = s_bias[3 * HDIM + j];
#pragma unroll
      for (int v = 0; v < 8; ++v) {
        float ig = sigmoidf_(acc[0][u][v] + bi);
        float fg = sigmoidf_(acc[1][u][v] + bf_);
        float gg = tanhf(acc[2][u][v] + bg);
        float og = sigmoidf_(acc[3][u][v] + bo);
        float c = fg * c_reg[u][v] + ig * gg;
        float h = og * tanhf(c);
        c_reg[u][v] = c;
        hnew[u][v] = h;
        int m = v + ((lane >> 4) << 3);
        s_act[m * (2 * HDIM) + HDIM + j] = (__bf16)h;
        s_act[m * (2 * HDIM) + j] = (__bf16)(h > 0.0f ? h : 0.0f);
      }
    }
    if (t == seq - 1) {   // final states (fp32, from registers)
#pragma unroll
      for (int u = 0; u < 2; ++u) {
        int j = wave * 32 + u * 16 + (lane & 15);
#pragma unroll
        for (int v = 0; v < 8; ++v) {
          int m = v + ((lane >> 4) << 3);
          hT[(size_t)(bbase + m) * HDIM + j] = hnew[u][v];
          cT[(size_t)(bbase + m) * HDIM + j] = c_reg[u][v];
        }
      }
    }
    __syncthreads();   // new act visible

    // y = h @ W_out^T + b_out  (N=128: waves 0..7, one 16-col tile each)
    if (wave < 8) {
      v8f y = vzero8();
      for (int kt = 0; kt < HDIM / 32; ++kt) {
        v16bf a = lds_a_frag(s_act + HDIM, 2 * HDIM, kt);
        y = wmma_bf16(a, g_b_frag(wout, wave * (HDIM / 32) + kt), y);
      }
      int o = wave * 16 + (lane & 15);
      float bo = b_out[o];
#pragma unroll
      for (int v = 0; v < 8; ++v) {
        int m = v + ((lane >> 4) << 3);
        dec_out[(size_t)(bbase + m) * seq * ODIM + (size_t)t * ODIM + o] = y[v] + bo;
      }
    }
  }
}

// ---------------------------------------------------------------------------
// num head: relu(lrelu(enc_out @ W_seq^T + b_seq) @ W_seq2^T + b2)
// 16 rows per WG, hidden never leaves registers; fused dot-reduction.
// ---------------------------------------------------------------------------
__global__ void __launch_bounds__(256)
num_kernel(const float* __restrict__ enc_out, const float* __restrict__ b_seq,
           const float* __restrict__ w2, const float* __restrict__ b2,
           const __bf16* __restrict__ wseq, float* __restrict__ num_out) {
  __shared__ __align__(32) __bf16 s_a[16 * LDIM];
  __shared__ float s_w2[HDIM], s_bs[HDIM], s_part[16];

  const int tid = threadIdx.x, wave = tid >> 5, lane = tid & 31;
  const size_t row0 = (size_t)blockIdx.x * 16;

  for (int i = tid; i < HDIM; i += 256) { s_w2[i] = w2[i]; s_bs[i] = b_seq[i]; }
  if (tid < 16) s_part[tid] = 0.0f;
  for (int idx = tid; idx < 16 * LDIM; idx += 256) {
    int m = idx >> 8, k = idx & (LDIM - 1);
    s_a[idx] = (__bf16)enc_out[(row0 + m) * LDIM + k];
  }
  __syncthreads();

  v8f acc[4] = {vzero8(), vzero8(), vzero8(), vzero8()};
  for (int kt = 0; kt < LDIM / 32; ++kt) {
    v16bf a = lds_a_frag(s_a, LDIM, kt);
#pragma unroll
    for (int u = 0; u < 4; ++u)
      acc[u] = wmma_bf16(a, g_b_frag(wseq, (wave * 4 + u) * (LDIM / 32) + kt), acc[u]);
  }

  float psum[8] = {0, 0, 0, 0, 0, 0, 0, 0};
#pragma unroll
  for (int u = 0; u < 4; ++u) {
    int n = wave * 64 + u * 16 + (lane & 15);
    float bn = s_bs[n], wn = s_w2[n];
#pragma unroll
    for (int v = 0; v < 8; ++v) psum[v] += lrelu_(acc[u][v] + bn) * wn;
  }
#pragma unroll
  for (int v = 0; v < 8; ++v) {
    float s = psum[v];
    for (int off = 1; off < 16; off <<= 1) s += __shfl_xor(s, off, 32);
    if ((lane & 15) == 0) atomicAdd(&s_part[v + ((lane >> 4) << 3)], s);
  }
  __syncthreads();
  if (tid < 16) {
    float val = s_part[tid] + b2[0];
    num_out[row0 + tid] = val > 0.0f ? val : 0.0f;
  }
}

// ---------------------------------------------------------------------------
// Host launcher
// ---------------------------------------------------------------------------
extern "C" void kernel_launch(void* const* d_in, const int* in_sizes, int n_in,
                              void* d_out, int out_size, void* d_ws, size_t ws_size,
                              hipStream_t stream) {
  const float* enc_out     = (const float*)d_in[0];
  const float* enc_hidden  = (const float*)d_in[1];
  const float* start_token = (const float*)d_in[2];
  const float* W_lat2hid   = (const float*)d_in[3];
  const float* b_lat2hid   = (const float*)d_in[4];
  const float* W_lat2hid2  = (const float*)d_in[5];
  const float* b_lat2hid2  = (const float*)d_in[6];
  const float* W_emb       = (const float*)d_in[7];
  const float* b_emb       = (const float*)d_in[8];
  const float* W_ih        = (const float*)d_in[9];
  const float* W_hh        = (const float*)d_in[10];
  const float* b_ih        = (const float*)d_in[11];
  const float* b_hh        = (const float*)d_in[12];
  const float* W_out       = (const float*)d_in[13];
  const float* b_out       = (const float*)d_in[14];
  const float* W_seq       = (const float*)d_in[15];
  const float* b_seq       = (const float*)d_in[16];
  const float* W_seq2      = (const float*)d_in[17];
  const float* b_seq2      = (const float*)d_in[18];

  // seq_len is only on device; derive it deterministically from out_size.
  int seq = (out_size - 2 * BDIM * HDIM - BDIM * TENC) / (BDIM * ODIM);
  if (seq < 1) seq = 1;

  // workspace: packed bf16 weight fragments
  const size_t N_WCAT = 128u * 32u * 512u;             // 2,097,152
  const size_t N_WOUT = 8u * 16u * 512u;               //    65,536
  const size_t N_WL2H = 32u * 8u * 512u;               //   131,072
  const size_t N_WSEQ = 32u * 8u * 512u;               //   131,072
  const size_t need = (N_WCAT + N_WOUT + 2 * N_WL2H + N_WSEQ) * sizeof(__bf16);
  if (ws_size < need) return;

  __bf16* wcat  = (__bf16*)d_ws;
  __bf16* wout  = wcat + N_WCAT;
  __bf16* wl2h  = wout + N_WOUT;
  __bf16* wl2h2 = wl2h + N_WL2H;
  __bf16* wseq  = wl2h2 + N_WL2H;

  pack_wcat_kernel<<<(int)(N_WCAT / 256), 256, 0, stream>>>(W_ih, W_hh, wcat);
  pack_b_kernel<<<(int)(N_WOUT / 256), 256, 0, stream>>>(W_out, wout, ODIM, HDIM);
  pack_b_kernel<<<(int)(N_WL2H / 256), 256, 0, stream>>>(W_lat2hid, wl2h, HDIM, LDIM);
  pack_b_kernel<<<(int)(N_WL2H / 256), 256, 0, stream>>>(W_lat2hid2, wl2h2, HDIM, LDIM);
  pack_b_kernel<<<(int)(N_WSEQ / 256), 256, 0, stream>>>(W_seq, wseq, HDIM, LDIM);

  float* out = (float*)d_out;
  float* dec = out;                                   // [B, seq, O]
  float* hT  = dec + (size_t)BDIM * seq * ODIM;       // [1, B, H]
  float* cT  = hT + (size_t)BDIM * HDIM;              // [1, B, H]
  float* num = cT + (size_t)BDIM * HDIM;              // [B, T_ENC, 1]

  decoder_kernel<<<BDIM / 16, 512, 0, stream>>>(
      enc_hidden, start_token, W_emb, b_emb, b_lat2hid, b_lat2hid2,
      b_ih, b_hh, b_out, wcat, wl2h, wl2h2, wout, seq, dec, hT, cT);

  num_kernel<<<(BDIM * TENC) / 16, 256, 0, stream>>>(
      enc_out, b_seq, W_seq2, b_seq2, wseq, num);
}
